// RefinedSampling_76381698392660
// MI455X (gfx1250) — matrix operand order repaired and checked by reading, not compiled, not run on hardware
//
#include <hip/hip_runtime.h>
#include <hip/hip_bf16.h>

typedef __attribute__((ext_vector_type(16))) _Float16 v16h;
typedef __attribute__((ext_vector_type(8)))  _Float16 v8h;
typedef __attribute__((ext_vector_type(8)))  float    v8f;

#define NPTS 8192
#define NB   8
#define NM   1024
#define NK   32
#define NC   128

// ---------------------------------------------------------------------------
// Kernel 0: convert weights to f16, pad w1_0's 131 input channels to 160.
// ---------------------------------------------------------------------------
__global__ void wcvt_kernel(const float* __restrict__ w10,
                            const float* __restrict__ w11,
                            const float* __restrict__ w12,
                            _Float16* __restrict__ h10,
                            _Float16* __restrict__ h11,
                            _Float16* __restrict__ h12)
{
    int t = blockIdx.x * 256 + threadIdx.x;
    if (t < 128 * 160) {
        int co = t / 160, ci = t % 160;
        h10[t] = (_Float16)(ci < 131 ? w10[co * 131 + ci] : 0.0f);
    } else if (t < 128 * 160 + 128 * 128) {
        int u = t - 128 * 160;
        h11[u] = (_Float16)w11[u];
    } else if (t < 128 * 160 + 128 * 128 + 256 * 128) {
        int u = t - (128 * 160 + 128 * 128);
        h12[u] = (_Float16)w12[u];
    }
}

// ---------------------------------------------------------------------------
// Kernel 1: farthest point sampling. One block per batch, 1024 threads,
// 8 points per thread held in registers. Serial over M steps.
// ---------------------------------------------------------------------------
__global__ __launch_bounds__(1024) void fps_kernel(const float* __restrict__ points,
                                                   int* __restrict__ idx_out,
                                                   float* __restrict__ cent)
{
    __shared__ float wv[32];
    __shared__ int   wi[32];
    __shared__ float sx, sy, sz;
    __shared__ int   sidx;

    const int b    = blockIdx.x;
    const int tid  = threadIdx.x;
    const int lane = tid & 31;
    const int wave = tid >> 5;
    const float* px = points + (size_t)(b * 3 + 0) * NPTS;
    const float* py = points + (size_t)(b * 3 + 1) * NPTS;
    const float* pz = points + (size_t)(b * 3 + 2) * NPTS;

    float x[8], y[8], z[8], md[8];
#pragma unroll
    for (int j = 0; j < 8; ++j) {
        int n = tid + j * 1024;
        x[j] = px[n]; y[j] = py[n]; z[j] = pz[n];
        md[j] = 1.0e10f;
    }
    if (tid == 0) { sidx = 0; sx = x[0]; sy = y[0]; sz = z[0]; }
    __syncthreads();

    for (int m = 0; m < NM; ++m) {
        int   sel = sidx;
        float lx = sx, ly = sy, lz = sz;
        if (tid == 0) {
            idx_out[b * NM + m] = sel;
            cent[((size_t)b * NM + m) * 3 + 0] = lx;
            cent[((size_t)b * NM + m) * 3 + 1] = ly;
            cent[((size_t)b * NM + m) * 3 + 2] = lz;
        }
        float bv = -1.0f; int bn = 0x7fffffff;
#pragma unroll
        for (int j = 0; j < 8; ++j) {
            float dx = x[j] - lx, dy = y[j] - ly, dz = z[j] - lz;
            float d = dx * dx + dy * dy + dz * dz;
            md[j] = fminf(md[j], d);
            int n = tid + j * 1024;
            if (md[j] > bv || (md[j] == bv && n < bn)) { bv = md[j]; bn = n; }
        }
        for (int off = 16; off > 0; off >>= 1) {   // wave32 reduce
            float ov = __shfl_down(bv, off);
            int   on = __shfl_down(bn, off);
            if (ov > bv || (ov == bv && on < bn)) { bv = ov; bn = on; }
        }
        if (lane == 0) { wv[wave] = bv; wi[wave] = bn; }
        __syncthreads();
        if (wave == 0) {
            float cv = wv[lane]; int cn = wi[lane];
            for (int off = 16; off > 0; off >>= 1) {
                float ov = __shfl_down(cv, off);
                int   on = __shfl_down(cn, off);
                if (ov > cv || (ov == cv && on < cn)) { cv = ov; cn = on; }
            }
            if (lane == 0) sidx = cn;
        }
        __syncthreads();
        int ns = sidx;
        if (tid == (ns & 1023)) {       // owner publishes new centroid coords
            int j = ns >> 10;
            sx = x[j]; sy = y[j]; sz = z[j];
        }
        __syncthreads();
    }
}

// ---------------------------------------------------------------------------
// Kernel 2: ball query. One wave32 per (b,m).
// ---------------------------------------------------------------------------
__global__ __launch_bounds__(256) void bq_kernel(const float* __restrict__ points,
                                                 const float* __restrict__ cent,
                                                 int* __restrict__ gidx)
{
    const int wib  = threadIdx.x >> 5;
    const int lane = threadIdx.x & 31;
    const int bm   = blockIdx.x * 8 + wib;
    const int b    = bm >> 10;
    const float r2 = 0.2f * 0.2f;
    const float* px = points + (size_t)(b * 3 + 0) * NPTS;
    const float* py = points + (size_t)(b * 3 + 1) * NPTS;
    const float* pz = points + (size_t)(b * 3 + 2) * NPTS;
    const float cx = cent[(size_t)bm * 3 + 0];
    const float cy = cent[(size_t)bm * 3 + 1];
    const float cz = cent[(size_t)bm * 3 + 2];
    int* gout = gidx + (size_t)bm * NK;

    int cnt = 0;
    int first = 0;
    for (int n0 = 0; n0 < NPTS && cnt < NK; n0 += 32) {
        int n = n0 + lane;
        float dx = px[n] - cx, dy = py[n] - cy, dz = pz[n] - cz;
        float d2 = dx * dx + dy * dy + dz * dz;
        bool pred = d2 <= r2;
        unsigned mask = (unsigned)(__ballot(pred) & 0xffffffffull);
        if (cnt == 0 && mask != 0u) first = n0 + __builtin_ctz(mask);
        int before = __popc(mask & ((1u << lane) - 1u));
        int slot = cnt + before;
        if (pred && slot < NK) gout[slot] = n;
        cnt += __popc(mask);
    }
    if (cnt < NK) {                 // centroid itself always qualifies -> cnt>=1
        int s = cnt + lane;
        if (s < NK) gout[s] = first;
    }
}

// ---------------------------------------------------------------------------
// WMMA layer: Y(cout x 32 pts) = relu(W * X + b), f16 in, f32 accumulate.
// Each wave owns whole m-tiles (16 output channels) and computes BOTH
// 16-point n-tiles with two accumulators, so one A(weight) fragment load
// feeds two back-to-back v_wmma issues (halves weight traffic + load waits).
// ---------------------------------------------------------------------------
__device__ __forceinline__ void run_layer(const _Float16* __restrict__ wA, int wstride,
                                          const float* __restrict__ bias,
                                          const _Float16* inL, int istride,
                                          _Float16* outL, int ostride,
                                          int cout, int kchunks,
                                          int wave, int lane)
{
    const int hi    = lane >> 4;
    const int lrow  = lane & 15;
    const int mtiles = cout >> 4;
    for (int mt = wave; mt < mtiles; mt += 4) {
        const int cobase = mt << 4;
        v8f acc0 = {};
        v8f acc1 = {};
        for (int kc = 0; kc < kchunks; ++kc) {
            // A fragment (shared by both n-tiles)
            const _Float16* ap = wA + (size_t)(cobase + lrow) * wstride + kc * 32 + hi * 16;
            v8h a0 = *(const v8h*)(ap);
            v8h a1 = *(const v8h*)(ap + 8);
            // B fragments for n-tile 0 (points 0..15) and 1 (points 16..31)
            const _Float16* bp0 = inL + (size_t)lrow * istride + kc * 32 + hi * 16;
            const _Float16* bp1 = bp0 + 16 * istride;
            v8h b00 = *(const v8h*)(bp0);
            v8h b01 = *(const v8h*)(bp0 + 8);
            v8h b10 = *(const v8h*)(bp1);
            v8h b11 = *(const v8h*)(bp1 + 8);
            v16h af, bf0, bf1;
#pragma unroll
            for (int e = 0; e < 8; ++e) {
                af[e]  = a0[e];  af[e + 8]  = a1[e];
                bf0[e] = b00[e]; bf0[e + 8] = b01[e];
                bf1[e] = b10[e]; bf1[e + 8] = b11[e];
            }
            acc0 = __builtin_amdgcn_wmma_f32_16x16x32_f16(false, af, false, bf0,
                                                          (short)0, acc0, false, false);
            acc1 = __builtin_amdgcn_wmma_f32_16x16x32_f16(false, af, false, bf1,
                                                          (short)0, acc1, false, false);
        }
        v8h o0, o1;
#pragma unroll
        for (int r = 0; r < 8; ++r) {
            float bb = bias[cobase + r + hi * 8];
            float v0 = acc0[r] + bb;
            float v1 = acc1[r] + bb;
            o0[r] = (_Float16)(v0 > 0.0f ? v0 : 0.0f);
            o1[r] = (_Float16)(v1 > 0.0f ? v1 : 0.0f);
        }
        _Float16* op = outL + (size_t)lrow * ostride + cobase + hi * 8;
        *(v8h*)(op)                 = o0;
        *(v8h*)(op + 16 * ostride)  = o1;
    }
}

// ---------------------------------------------------------------------------
// Kernel 3: one block (4 waves) per (b,m) group: gather -> 3 WMMA layers ->
// max over K -> 256->128->3 head -> out = cent + x.
// ---------------------------------------------------------------------------
__global__ __launch_bounds__(128) void mlp_kernel(const float* __restrict__ points,
                                                  const float* __restrict__ feats,
                                                  const int*   __restrict__ gidx,
                                                  const float* __restrict__ cent,
                                                  const _Float16* __restrict__ h10, const float* __restrict__ b10,
                                                  const _Float16* __restrict__ h11, const float* __restrict__ b11,
                                                  const _Float16* __restrict__ h12, const float* __restrict__ b12,
                                                  const float* __restrict__ w20, const float* __restrict__ b20,
                                                  const float* __restrict__ w21, const float* __restrict__ b21,
                                                  float* __restrict__ out)
{
    __shared__ __attribute__((aligned(16))) _Float16 sA[32 * 160];
    __shared__ __attribute__((aligned(16))) _Float16 sB[32 * 256];
    __shared__ float red[256];
    __shared__ float l4[128];
    __shared__ int   gk[32];
    __shared__ float cc[3];

    const int tid  = threadIdx.x;
    const int wave = tid >> 5;
    const int lane = tid & 31;
    const int bm   = blockIdx.x;
    const int b    = bm >> 10;
    const int m    = bm & (NM - 1);

    if (tid < 32) gk[tid] = gidx[(size_t)bm * NK + tid];
    if (tid < 3)  cc[tid] = cent[(size_t)bm * 3 + tid];
    __syncthreads();

    // gather: 32 points x 160 channels f16 (0..2 = xyz - cent, 3..130 = feat, rest 0)
    {
        const int k  = tid >> 2;
        const int c0 = (tid & 3) * 40;
        const int gi = gk[k];
        const float* pb = points + (size_t)b * 3 * NPTS;
        const float* fb = feats  + (size_t)b * NC * NPTS;
#pragma unroll 4
        for (int c = c0; c < c0 + 40; ++c) {
            float v;
            if (c < 3)        v = pb[(size_t)c * NPTS + gi] - cc[c];
            else if (c < 131) v = fb[(size_t)(c - 3) * NPTS + gi];
            else              v = 0.0f;
            sA[k * 160 + c] = (_Float16)v;
        }
    }
    __syncthreads();

    run_layer(h10, 160, b10, sA, 160, sB, 256, 128, 5, wave, lane);  // 131(->160) -> 128
    __syncthreads();
    run_layer(h11, 128, b11, sB, 256, sA, 160, 128, 4, wave, lane);  // 128 -> 128
    __syncthreads();
    run_layer(h12, 128, b12, sA, 160, sB, 256, 256, 4, wave, lane);  // 128 -> 256
    __syncthreads();

    // max over the 32 neighbors -> red[256]
#pragma unroll
    for (int t = 0; t < 2; ++t) {
        int co = tid + t * 128;
        float mx = 0.0f;                       // post-relu values are >= 0
        #pragma unroll 8
        for (int k = 0; k < 32; ++k) mx = fmaxf(mx, (float)sB[k * 256 + co]);
        red[co] = mx;
    }
    __syncthreads();

    // head layer 1: 256 -> 128, relu
    {
        float acc = b20[tid];
        const float* wr = w20 + (size_t)tid * 256;
        #pragma unroll 8
        for (int c = 0; c < 256; ++c) acc += wr[c] * red[c];
        l4[tid] = acc > 0.0f ? acc : 0.0f;
    }
    __syncthreads();

    // head layer 2: 128 -> 3, relu, + centroid; out[b][d][m]
    if (tid < 3) {
        float acc = b21[tid];
        const float* wr = w21 + (size_t)tid * 128;
        for (int c = 0; c < 128; ++c) acc += wr[c] * l4[c];
        acc = acc > 0.0f ? acc : 0.0f;
        out[((size_t)b * 3 + tid) * NM + m] = cc[tid] + acc;
    }
}

// ---------------------------------------------------------------------------
extern "C" void kernel_launch(void* const* d_in, const int* in_sizes, int n_in,
                              void* d_out, int out_size, void* d_ws, size_t ws_size,
                              hipStream_t stream)
{
    (void)in_sizes; (void)n_in; (void)out_size; (void)ws_size;

    const float* points = (const float*)d_in[0];
    const float* feats  = (const float*)d_in[1];
    const float* w10 = (const float*)d_in[2];
    const float* b10 = (const float*)d_in[3];
    const float* w11 = (const float*)d_in[4];
    const float* b11 = (const float*)d_in[5];
    const float* w12 = (const float*)d_in[6];
    const float* b12 = (const float*)d_in[7];
    const float* w20 = (const float*)d_in[8];
    const float* b20 = (const float*)d_in[9];
    const float* w21 = (const float*)d_in[10];
    const float* b21 = (const float*)d_in[11];
    float* out = (float*)d_out;

    char* ws = (char*)d_ws;
    int*      idxp  = (int*)ws;        ws += (size_t)NB * NM * 4;            // 32 KB
    float*    centp = (float*)ws;      ws += (size_t)NB * NM * 3 * 4;        // 96 KB
    int*      gidxp = (int*)ws;        ws += (size_t)NB * NM * NK * 4;       // 1 MB
    _Float16* h10   = (_Float16*)ws;   ws += (size_t)128 * 160 * 2;
    _Float16* h11   = (_Float16*)ws;   ws += (size_t)128 * 128 * 2;
    _Float16* h12   = (_Float16*)ws;   ws += (size_t)256 * 128 * 2;

    const int wtot = 128 * 160 + 128 * 128 + 256 * 128;   // 69632
    wcvt_kernel<<<(wtot + 255) / 256, 256, 0, stream>>>(w10, w11, w12, h10, h11, h12);
    fps_kernel<<<NB, 1024, 0, stream>>>(points, idxp, centp);
    bq_kernel<<<(NB * NM) / 8, 256, 0, stream>>>(points, centp, gidxp);
    mlp_kernel<<<NB * NM, 128, 0, stream>>>(points, feats, gidxp, centp,
                                            h10, b10, h11, b11, h12, b12,
                                            w20, b20, w21, b21, out);
}
